// SessionGraph_64123861729506
// MI455X (gfx1250) — matrix operand order repaired and checked by reading, not compiled, not run on hardware
//
#include <hip/hip_runtime.h>

#define N_NODE 40000
#define DIM 128
#define N_REL 8
#define N_EDGES 640000
#define BS 128
#define SQ 50
#define PER_REL (N_EDGES / N_REL)   // 80000

typedef __attribute__((ext_vector_type(16))) __bf16 v16bf;
typedef __attribute__((ext_vector_type(8)))  float  v8f;

union FragBF {
  v16bf v;
  struct { uint4 lo, hi; } q;
};

__device__ __forceinline__ unsigned short f2bf(float f) {
  unsigned int u = __float_as_uint(f);
  u += 0x7FFFu + ((u >> 16) & 1u);          // round-to-nearest-even
  return (unsigned short)(u >> 16);
}

__device__ __forceinline__ void atomicMaxF32(float* addr, float val) {
  // monotonic trick; correct with -inf init
  int iv = __float_as_int(val);
  if (iv >= 0) atomicMax((int*)addr, iv);
  else         atomicMin((unsigned int*)addr, (unsigned int)iv);
}

// ---------------------------------------------------------------- init
__global__ void __launch_bounds__(256)
init_neg_inf_kernel(float* __restrict__ p, int n) {
  int i = blockIdx.x * 256 + threadIdx.x;
  if (i < n) p[i] = -__builtin_inff();
}

// relsum[r] = sum_d r_embed_w[r,d]
__global__ void __launch_bounds__(256)
relsum_kernel(const float* __restrict__ r_embed_w, float* __restrict__ relsum) {
  int rel = threadIdx.x >> 5, lane = threadIdx.x & 31;
  const float* row = r_embed_w + rel * DIM;
  float s = row[lane] + row[lane + 32] + row[lane + 64] + row[lane + 96];
  for (int off = 16; off; off >>= 1) s += __shfl_xor(s, off, 32);
  if (lane == 0) relsum[rel] = s;
}

// ---------------------------------------------------------------- spmm
// out[h[e],:] += v[e] * x[t[e],:]   (one wave per edge, 4 floats/lane)
__global__ void __launch_bounds__(256)
spmm_kernel(const float* __restrict__ v, const int* __restrict__ rows,
            const int* __restrict__ cols, const float* __restrict__ x,
            float* __restrict__ out) {
  int gid = blockIdx.x * 256 + threadIdx.x;
  int e = gid >> 5, lane = gid & 31;
  float s = v[e];
  int t = cols[e], h = rows[e];
  const float4 xv = *(const float4*)(x + (size_t)t * DIM + lane * 4);
  float* o = out + (size_t)h * DIM + lane * 4;
  atomicAdd(o + 0, xv.x * s);
  atomicAdd(o + 1, xv.y * s);
  atomicAdd(o + 2, xv.z * s);
  atomicAdd(o + 3, xv.w * s);
}

// ---------------------------------------------------------------- update_attention (WMMA)
// vnew[e] = sum_d tanh((ego[t[e]] @ W_r)_d) * (ego[h[e]] @ W_r)_d + relsum[r]
// block: 256 threads = 8 waves, 16 edges/wave -> 128 edges/block, 625 blocks/relation
__global__ void __launch_bounds__(256)
attn_kernel(const float* __restrict__ ego,
            const int* __restrict__ hlist, const int* __restrict__ tlist,
            const float* __restrict__ transM, const float* __restrict__ relsum,
            float* __restrict__ vnew) {
  extern __shared__ unsigned short smem[];
  unsigned short* Wt = smem;                        // [128][136] W transposed, bf16
  unsigned short* Ah = smem + 128 * 136;            // 8 waves x [16][136]
  unsigned short* At = Ah + 8 * 16 * 136;

  const int tid  = threadIdx.x;
  const int w    = tid >> 5;
  const int lane = tid & 31;
  const int l16  = lane & 15;
  const int hi   = lane >> 4;

  const int blockBase = blockIdx.x * 128;
  const int rel  = blockBase / PER_REL;
  const int base = blockBase + w * 16;

  // stage W_r^T into LDS as bf16 (coalesced global reads)
  const float* W = transM + (size_t)rel * (DIM * DIM);
  for (int idx = tid; idx < DIM * DIM; idx += 256) {
    int k = idx >> 7, n = idx & 127;
    Wt[n * 136 + k] = f2bf(W[idx]);
  }

  // stage gathered A rows (16 h-rows + 16 t-rows per wave) as bf16
  unsigned short* ah = Ah + w * (16 * 136);
  unsigned short* at = At + w * (16 * 136);
  for (int m = 0; m < 16; ++m) {
    int hh = hlist[base + m];
    int tt = tlist[base + m];
    float4 fh = *(const float4*)(ego + (size_t)hh * DIM + lane * 4);
    float4 ft = *(const float4*)(ego + (size_t)tt * DIM + lane * 4);
    unsigned short* ra = ah + m * 136 + lane * 4;
    unsigned short* rt = at + m * 136 + lane * 4;
    ra[0] = f2bf(fh.x); ra[1] = f2bf(fh.y); ra[2] = f2bf(fh.z); ra[3] = f2bf(fh.w);
    rt[0] = f2bf(ft.x); rt[1] = f2bf(ft.y); rt[2] = f2bf(ft.z); rt[3] = f2bf(ft.w);
  }
  __syncthreads();

  float lsum[8] = {0.f, 0.f, 0.f, 0.f, 0.f, 0.f, 0.f, 0.f};

  const unsigned short* arow = ah + l16 * 136;
  const unsigned short* trow = at + l16 * 136;

#pragma unroll
  for (int nt = 0; nt < 8; ++nt) {
    v8f accH = {0.f, 0.f, 0.f, 0.f, 0.f, 0.f, 0.f, 0.f};
    v8f accT = {0.f, 0.f, 0.f, 0.f, 0.f, 0.f, 0.f, 0.f};
    const unsigned short* brow = Wt + (nt * 16 + l16) * 136;
#pragma unroll
    for (int ks = 0; ks < 4; ++ks) {
      FragBF fa, fb, fc;
      const int ca = ks * 32 + hi * 8;       // A: elems 0..7 -> K=ca.., 8..15 -> K=ca+16..
      fa.q.lo = *(const uint4*)(arow + ca);
      fa.q.hi = *(const uint4*)(arow + ca + 16);
      fc.q.lo = *(const uint4*)(trow + ca);
      fc.q.hi = *(const uint4*)(trow + ca + 16);
      const int cb = ks * 32 + hi * 8;       // B: elems 0..7 -> K=cb.., 8..15 -> K=cb+16.., N=l16 (from W^T rows)
      fb.q.lo = *(const uint4*)(brow + cb);
      fb.q.hi = *(const uint4*)(brow + cb + 16);
      accH = __builtin_amdgcn_wmma_f32_16x16x32_bf16(false, fa.v, false, fb.v,
                                                     (short)0, accH, false, false);
      accT = __builtin_amdgcn_wmma_f32_16x16x32_bf16(false, fc.v, false, fb.v,
                                                     (short)0, accT, false, false);
    }
#pragma unroll
    for (int r = 0; r < 8; ++r)
      lsum[r] += tanhf(accT[r]) * accH[r];   // lane holds D[m=r+8*hi, n=l16]
  }

  // reduce over the 16 n-columns within each half-wave
#pragma unroll
  for (int r = 0; r < 8; ++r) {
    float s = lsum[r];
    s += __shfl_xor(s, 8, 16);
    s += __shfl_xor(s, 4, 16);
    s += __shfl_xor(s, 2, 16);
    s += __shfl_xor(s, 1, 16);
    lsum[r] = s;
  }
  if (l16 == 0) {
    float rs = relsum[rel];
#pragma unroll
    for (int r = 0; r < 8; ++r)
      vnew[base + hi * 8 + r] = lsum[r] + rs;
  }
}

// ---------------------------------------------------------------- sparse row softmax
__global__ void __launch_bounds__(256)
seg_max_kernel(const float* __restrict__ vnew, const int* __restrict__ rows,
               float* __restrict__ rowmax) {
  int e = blockIdx.x * 256 + threadIdx.x;
  atomicMaxF32(&rowmax[rows[e]], vnew[e]);
}

__global__ void __launch_bounds__(256)
exp_sum_kernel(const float* __restrict__ vnew, const int* __restrict__ rows,
               const float* __restrict__ rowmax, float* __restrict__ vbuf,
               float* __restrict__ rowsum) {
  int e = blockIdx.x * 256 + threadIdx.x;
  int h = rows[e];
  float ev = expf(vnew[e] - rowmax[h]);
  vbuf[e] = ev;
  atomicAdd(&rowsum[h], ev);
}

__global__ void __launch_bounds__(256)
norm_kernel(float* __restrict__ vbuf, const int* __restrict__ rows,
            const float* __restrict__ rowsum) {
  int e = blockIdx.x * 256 + threadIdx.x;
  vbuf[e] = vbuf[e] / rowsum[rows[e]];
}

// ---------------------------------------------------------------- gathers
// node_out[row,:] = embedding[idx]; kg_out[row,:] = 0.5*(ego1+ego2)[idx]
__global__ void __launch_bounds__(256)
gather_kernel(const int* __restrict__ inputs, const float* __restrict__ emb,
              const float* __restrict__ e1, const float* __restrict__ e2,
              float* __restrict__ node_out, float* __restrict__ kg_out) {
  int row = blockIdx.x * 8 + (threadIdx.x >> 5);
  int lane = threadIdx.x & 31;
  int idx = inputs[row];
  size_t src = (size_t)idx * DIM + lane * 4;
  size_t dst = (size_t)row * DIM + lane * 4;
  float4 nv = *(const float4*)(emb + src);
  float4 a  = *(const float4*)(e1 + src);
  float4 b  = *(const float4*)(e2 + src);
  *(float4*)(node_out + dst) = nv;
  float4 kg = {0.5f * (a.x + b.x), 0.5f * (a.y + b.y),
               0.5f * (a.z + b.z), 0.5f * (a.w + b.w)};
  *(float4*)(kg_out + dst) = kg;
}

// ---------------------------------------------------------------- local_agg
// one block per batch b; 8 waves; rows i strided by wave; 4 dims/lane
__global__ void __launch_bounds__(256)
local_agg_kernel(const float* __restrict__ src, const int* __restrict__ A,
                 const float* __restrict__ a0, const float* __restrict__ a1,
                 const float* __restrict__ a2, const float* __restrict__ a3,
                 float* __restrict__ out, int accumulate) {
  __shared__ float hs[SQ * DIM];
  const int b = blockIdx.x;
  const int tid = threadIdx.x;
  for (int i4 = tid; i4 < SQ * DIM / 4; i4 += 256)
    *(float4*)&hs[i4 * 4] = *(const float4*)(src + (size_t)b * SQ * DIM + i4 * 4);
  __syncthreads();

  const int w = tid >> 5, lane = tid & 31;
  const float4 A0 = *(const float4*)(a0 + lane * 4);
  const float4 A1 = *(const float4*)(a1 + lane * 4);
  const float4 A2 = *(const float4*)(a2 + lane * 4);
  const float4 A3 = *(const float4*)(a3 + lane * 4);
  const int* adj = A + b * SQ * SQ;

  for (int i = w; i < SQ; i += 8) {
    float4 hv = *(float4*)&hs[i * DIM + lane * 4];
    float lg[SQ];
    float mx = -3.0e38f;
#pragma unroll
    for (int j = 0; j < SQ; ++j) {
      float4 hj = *(float4*)&hs[j * DIM + lane * 4];
      float4 p = {hv.x * hj.x, hv.y * hj.y, hv.z * hj.z, hv.w * hj.w};
      p.x = p.x > 0.f ? p.x : 0.2f * p.x;
      p.y = p.y > 0.f ? p.y : 0.2f * p.y;
      p.z = p.z > 0.f ? p.z : 0.2f * p.z;
      p.w = p.w > 0.f ? p.w : 0.2f * p.w;
      float s0 = p.x * A0.x + p.y * A0.y + p.z * A0.z + p.w * A0.w;
      float s1 = p.x * A1.x + p.y * A1.y + p.z * A1.z + p.w * A1.w;
      float s2 = p.x * A2.x + p.y * A2.y + p.z * A2.z + p.w * A2.w;
      float s3 = p.x * A3.x + p.y * A3.y + p.z * A3.z + p.w * A3.w;
      for (int off = 16; off; off >>= 1) {
        s0 += __shfl_xor(s0, off, 32);
        s1 += __shfl_xor(s1, off, 32);
        s2 += __shfl_xor(s2, off, 32);
        s3 += __shfl_xor(s3, off, 32);
      }
      int code = adj[i * SQ + j];
      float lv = -9.0e15f;
      if (code == 1) lv = s0;
      else if (code == 2) lv = s1;
      else if (code == 3) lv = s2;
      else if (code == 4) lv = s3;
      lg[j] = lv;
      mx = fmaxf(mx, lv);
    }
    float denom = 0.f;
    float4 acc = {0.f, 0.f, 0.f, 0.f};
#pragma unroll
    for (int j = 0; j < SQ; ++j) {
      float ev = expf(lg[j] - mx);
      denom += ev;
      float4 hj = *(float4*)&hs[j * DIM + lane * 4];
      acc.x += ev * hj.x; acc.y += ev * hj.y;
      acc.z += ev * hj.z; acc.w += ev * hj.w;
    }
    float inv = 1.f / denom;
    float4 res = {acc.x * inv, acc.y * inv, acc.z * inv, acc.w * inv};
    float* op = out + (size_t)b * SQ * DIM + i * DIM + lane * 4;
    if (accumulate) {
      res.x += op[0]; res.y += op[1]; res.z += op[2]; res.w += op[3];
    }
    *(float4*)op = res;
  }
}

// ---------------------------------------------------------------- launch
extern "C" void kernel_launch(void* const* d_in, const int* in_sizes, int n_in,
                              void* d_out, int out_size, void* d_ws, size_t ws_size,
                              hipStream_t stream) {
  const int*   inputs    = (const int*)d_in[0];
  const int*   A         = (const int*)d_in[1];
  // d_in[2] = mask_item (unused in forward)
  const float* embedding = (const float*)d_in[3];
  const float* r_embed_w = (const float*)d_in[4];
  const float* trans_M   = (const float*)d_in[5];
  const float* a0        = (const float*)d_in[6];
  const float* a1        = (const float*)d_in[7];
  const float* a2        = (const float*)d_in[8];
  const float* a3        = (const float*)d_in[9];
  const float* a_in      = (const float*)d_in[10];
  const int*   hlist     = (const int*)d_in[11];
  const int*   tlist     = (const int*)d_in[12];
  float* out = (float*)d_out;

  float* ws     = (float*)d_ws;
  float* ego1   = ws;                                   // 5,120,000
  float* ego2   = ego1 + (size_t)N_NODE * DIM;          // 5,120,000
  float* vnew   = ego2 + (size_t)N_NODE * DIM;          // 640,000
  float* vbuf   = vnew + N_EDGES;                       // 640,000
  float* rowmax = vbuf + N_EDGES;                       // 40,000
  float* rowsum = rowmax + N_NODE;                      // 40,000
  float* relsum = rowsum + N_NODE;                      // 8
  float* kgg    = relsum + 8;                           // 819,200

  hipMemsetAsync(ego1, 0, (size_t)N_NODE * DIM * sizeof(float), stream);
  hipMemsetAsync(ego2, 0, (size_t)N_NODE * DIM * sizeof(float), stream);
  hipMemsetAsync(rowsum, 0, (size_t)N_NODE * sizeof(float), stream);
  init_neg_inf_kernel<<<(N_NODE + 255) / 256, 256, 0, stream>>>(rowmax, N_NODE);
  relsum_kernel<<<1, 256, 0, stream>>>(r_embed_w, relsum);

  // hop 0: ego1 = spmm(a_in, h, t, embedding)
  spmm_kernel<<<N_EDGES * 32 / 256, 256, 0, stream>>>(a_in, hlist, tlist, embedding, ego1);

  // attention logits per edge (WMMA)
  const unsigned smem = (128 * 136 + 2 * 8 * 16 * 136) * sizeof(unsigned short); // 104448 B
  attn_kernel<<<N_EDGES / 128, 256, smem, stream>>>(ego1, hlist, tlist, trans_M, relsum, vnew);

  // row-wise softmax over h
  seg_max_kernel<<<N_EDGES / 256, 256, 0, stream>>>(vnew, hlist, rowmax);
  exp_sum_kernel<<<N_EDGES / 256, 256, 0, stream>>>(vnew, hlist, rowmax, vbuf, rowsum);
  norm_kernel<<<N_EDGES / 256, 256, 0, stream>>>(vbuf, hlist, rowsum);

  // hop 1: ego2 = spmm(softmax_v, h, t, ego1)
  spmm_kernel<<<N_EDGES * 32 / 256, 256, 0, stream>>>(vbuf, hlist, tlist, ego1, ego2);

  // gathers: node_embeds -> second output half; kg gathered rows -> ws
  float* node_out = out + (size_t)BS * SQ * DIM;        // second output
  gather_kernel<<<BS * SQ / 8, 256, 0, stream>>>(inputs, embedding, ego1, ego2,
                                                 node_out, kgg);

  // hidden = local_agg(node_embeds) + local_agg(kg_embeds)
  local_agg_kernel<<<BS, 256, 0, stream>>>(node_out, A, a0, a1, a2, a3, out, 0);
  local_agg_kernel<<<BS, 256, 0, stream>>>(kgg, A, a0, a1, a2, a3, out, 1);
}
